// FineGrained_80642305950046
// MI455X (gfx1250) — matrix operand order repaired and checked by reading, not compiled, not run on hardware
//
#include <hip/hip_runtime.h>
#include <hip/hip_bf16.h>

typedef __attribute__((ext_vector_type(16))) _Float16 v16h;
typedef __attribute__((ext_vector_type(8)))  _Float16 v8h;
typedef __attribute__((ext_vector_type(8)))  float    v8f;

#define N_  8
#define C_  256
#define H_  56
#define W_  56
#define HW_ (H_ * W_)          // 3136
#define RB_ 64                 // square register-block edge
#define RR_ (HW_ / RB_)        // 49 regions per dim
#define REGIONS_ (RR_ * RR_)   // 2401 wave-jobs per (pair, n)
#define PACK_BYTES ((size_t)N_ * HW_ * C_ * 2)   // 12,845,056 per tensor
#define CENT_ELEMS (N_ * HW_)

// ---------------------------------------------------------------------------
// Kernel 1: L2-normalize over C and pack [N,C,H,W] f32 -> [N,HW,C] f16
// Block: 32 (hw) x 8 (channel-groups). Each cg-thread sums 32 channels.
// ---------------------------------------------------------------------------
__global__ __launch_bounds__(256) void norm_pack_kernel(
    const float* __restrict__ in0, const float* __restrict__ in1,
    const float* __restrict__ in2, const float* __restrict__ in3,
    _Float16* __restrict__ p0, _Float16* __restrict__ p1,
    _Float16* __restrict__ p2, _Float16* __restrict__ p3)
{
  const float* in; _Float16* out;
  switch (blockIdx.z) {
    case 0:  in = in0; out = p0; break;
    case 1:  in = in1; out = p1; break;
    case 2:  in = in2; out = p2; break;
    default: in = in3; out = p3; break;
  }
  const int n   = blockIdx.y;
  const int hw  = blockIdx.x * 32 + threadIdx.x;  // 0..HW-1
  const int cg  = threadIdx.y;                    // 0..7, each covers 32 channels

  const float* base = in + ((size_t)n * C_ + cg * 32) * HW_ + hw;
  float vals[32];
  float s = 0.f;
  #pragma unroll
  for (int cc = 0; cc < 32; ++cc) {
    float v = base[(size_t)cc * HW_];
    vals[cc] = v;
    s += v * v;
  }

  __shared__ float part[8][32];
  __shared__ float invn[32];
  part[cg][threadIdx.x] = s;
  __syncthreads();
  if (cg == 0) {
    float t = 0.f;
    #pragma unroll
    for (int g = 0; g < 8; ++g) t += part[g][threadIdx.x];
    invn[threadIdx.x] = 1.0f / fmaxf(sqrtf(t), 1e-12f);
  }
  __syncthreads();
  const float inv = invn[threadIdx.x];

  _Float16* op = out + ((size_t)n * HW_ + hw) * C_ + cg * 32;
  #pragma unroll
  for (int cc = 0; cc < 32; ++cc) op[cc] = (_Float16)(vals[cc] * inv);
}

// ---------------------------------------------------------------------------
// Kernel 2: per-position crop-space centers, per-batch radius^2, zero accums
// ---------------------------------------------------------------------------
__global__ __launch_bounds__(256) void setup_kernel(
    const float* __restrict__ coord1, const float* __restrict__ coord2,
    float* __restrict__ c1x, float* __restrict__ c1y,
    float* __restrict__ c2x, float* __restrict__ c2y,
    float* __restrict__ r2, float* __restrict__ acc)
{
  const int idx = blockIdx.x * 256 + threadIdx.x;
  if (idx < 2 * N_ * 2) acc[idx] = 0.f;
  if (idx < N_) {
    const float* a = coord1 + idx * 4;
    const float* b = coord2 + idx * 4;
    float qbw = (a[2] - a[0]) / W_, qbh = (a[3] - a[1]) / H_;
    float kbw = (b[2] - b[0]) / W_, kbh = (b[3] - b[1]) / H_;
    float d1 = qbw * qbw + qbh * qbh;
    float d2 = kbw * kbw + kbh * kbh;
    r2[idx] = 0.49f * fmaxf(d1, d2);   // (POS_RADIUS * max_bin_diag)^2
  }
  if (idx < N_ * HW_) {
    const int n = idx / HW_, hw = idx - n * HW_;
    const int ix = hw % W_, iy = hw / W_;
    const float* a = coord1 + n * 4;
    const float* b = coord2 + n * 4;
    c1x[idx] = (ix + 0.5f) * ((a[2] - a[0]) / W_) + a[0];
    c1y[idx] = (iy + 0.5f) * ((a[3] - a[1]) / H_) + a[1];
    c2x[idx] = (ix + 0.5f) * ((b[2] - b[0]) / W_) + b[0];
    c2y[idx] = (iy + 0.5f) * ((b[3] - b[1]) / H_) + b[1];
  }
}

// ---------------------------------------------------------------------------
// Kernel 3: register-blocked WMMA GEMM + masked reduction.
// Each wave owns a 64x64 output region: 4 M-subtiles x 4 N-subtiles,
// 16 v8f accumulators. Per K-step: 4 A-frags + 4 B-frags (16 b128 loads)
// feed 16 WMMAs -> 1 b128 load per WMMA, 32 FLOP/byte from L2.
// K = 256 = 8 steps -> 128 WMMAs per wave.
// ---------------------------------------------------------------------------
__global__ __launch_bounds__(256) void wmma_loss_kernel(
    const _Float16* __restrict__ q1p, const _Float16* __restrict__ q2p,
    const _Float16* __restrict__ k1p, const _Float16* __restrict__ k2p,
    const float* __restrict__ c1x, const float* __restrict__ c1y,
    const float* __restrict__ c2x, const float* __restrict__ c2y,
    const float* __restrict__ r2arr, float* __restrict__ acc)
{
  const int pair = blockIdx.z;
  const int n    = blockIdx.y;
  const int wave = threadIdx.x >> 5;
  const int lane = threadIdx.x & 31;
  const int reg  = blockIdx.x * 8 + wave;   // wave-uniform region id

  __shared__ float wnum[8], wcnt[8];

  float num = 0.f, cnt = 0.f;

  if (reg < REGIONS_) {   // uniform per wave -> EXEC all-1s inside
    const int rm = reg / RR_;
    const int rn = reg - rm * RR_;
    const int i0 = rm * RB_;     // 64 output rows
    const int j0 = rn * RB_;     // 64 output cols

    // pair 0: loss(q1,k2,coord1,coord2); pair 1: loss(q2,k1,coord2,coord1)
    const _Float16* qp = pair ? q2p : q1p;
    const _Float16* kp = pair ? k1p : k2p;
    const float* qx = pair ? c2x : c1x;
    const float* qy = pair ? c2y : c1y;
    const float* kx = pair ? c1x : c2x;
    const float* ky = pair ? c1y : c2y;

    const int l15   = lane & 15;
    const int ahalf = (lane < 16) ? 0 : 16;   // A: K-offset of lane half
    const int bhalf = (lane < 16) ? 0 : 16;   // B: K-offset of lane half
    const _Float16* arow = qp + ((size_t)n * HW_ + i0 + l15) * C_;
    const _Float16* brow = kp + ((size_t)n * HW_ + j0 + l15) * C_;

    v8f cacc[4][4];
    #pragma unroll
    for (int mt = 0; mt < 4; ++mt)
      #pragma unroll
      for (int nt = 0; nt < 4; ++nt)
        cacc[mt][nt] = (v8f){};

    #pragma unroll
    for (int kb = 0; kb < 8; ++kb) {
      const int k0 = kb * 32;
      // A fragments: elems 0..7 = K[k0+ahalf..],
      // elems 8..15 = K[k0+ahalf+8..] (16 contiguous K per lane half, matches B)
      v16h a[4];
      #pragma unroll
      for (int mt = 0; mt < 4; ++mt) {
        const _Float16* ar = arow + (size_t)mt * 16 * C_;
        v8h lo = *(const v8h*)(ar + k0 + ahalf);
        v8h hi = *(const v8h*)(ar + k0 + ahalf + 8);
        a[mt] = __builtin_shufflevector(lo, hi,
            0, 1, 2, 3, 4, 5, 6, 7, 8, 9, 10, 11, 12, 13, 14, 15);
      }
      // B fragments: lane half covers 16 contiguous K values of column j
      v16h b[4];
      #pragma unroll
      for (int nt = 0; nt < 4; ++nt)
        b[nt] = *(const v16h*)(brow + (size_t)nt * 16 * C_ + k0 + bhalf);

      #pragma unroll
      for (int mt = 0; mt < 4; ++mt)
        #pragma unroll
        for (int nt = 0; nt < 4; ++nt)
          cacc[mt][nt] = __builtin_amdgcn_wmma_f32_16x16x32_f16(
              false, a[mt], false, b[nt], (short)0, cacc[mt][nt],
              false, false);
    }

    // ---- masked reduction over the 64x64 region ----
    // C/D layout: col = j0 + nt*16 + (lane&15);
    //             elem v -> row = i0 + mt*16 + v + (lane>=16 ? 8 : 0)
    const size_t cb = (size_t)n * HW_;
    const float rr = r2arr[n];
    const int moff = (lane < 16) ? 0 : 8;

    float kxv[4], kyv[4];
    #pragma unroll
    for (int nt = 0; nt < 4; ++nt) {
      kxv[nt] = kx[cb + j0 + nt * 16 + l15];
      kyv[nt] = ky[cb + j0 + nt * 16 + l15];
    }

    #pragma unroll
    for (int mt = 0; mt < 4; ++mt) {
      #pragma unroll
      for (int v = 0; v < 8; ++v) {
        const int i = i0 + mt * 16 + moff + v;
        const float qxx = qx[cb + i];
        const float qyy = qy[cb + i];
        #pragma unroll
        for (int nt = 0; nt < 4; ++nt) {
          const float dx = qxx - kxv[nt];
          const float dy = qyy - kyv[nt];
          const bool m = (dx * dx + dy * dy) < rr;
          num += m ? cacc[mt][nt][v] : 0.f;
          cnt += m ? 1.f : 0.f;
        }
      }
    }
  }

  // wave32 reduction
  #pragma unroll
  for (int s = 16; s; s >>= 1) {
    num += __shfl_xor(num, s, 32);
    cnt += __shfl_xor(cnt, s, 32);
  }

  if (lane == 0) { wnum[wave] = num; wcnt[wave] = cnt; }
  __syncthreads();
  if (threadIdx.x == 0) {
    float tn = 0.f, tc = 0.f;
    #pragma unroll
    for (int w = 0; w < 8; ++w) { tn += wnum[w]; tc += wcnt[w]; }
    atomicAdd(&acc[(pair * N_ + n) * 2 + 0], tn);
    atomicAdd(&acc[(pair * N_ + n) * 2 + 1], tc);
  }
}

// ---------------------------------------------------------------------------
// Kernel 4: final scalar
// ---------------------------------------------------------------------------
__global__ void finalize_kernel(const float* __restrict__ acc,
                                float* __restrict__ out)
{
  if (threadIdx.x == 0 && blockIdx.x == 0) {
    float s = 0.f;
    for (int p = 0; p < 2; ++p)
      for (int n = 0; n < N_; ++n)
        s += acc[(p * N_ + n) * 2 + 0] / (acc[(p * N_ + n) * 2 + 1] + 1e-6f);
    out[0] = -2.0f * s / (float)N_;
  }
}

// ---------------------------------------------------------------------------
extern "C" void kernel_launch(void* const* d_in, const int* in_sizes, int n_in,
                              void* d_out, int out_size, void* d_ws, size_t ws_size,
                              hipStream_t stream)
{
  (void)in_sizes; (void)n_in; (void)out_size; (void)ws_size;
  const float* pred1  = (const float*)d_in[0];
  const float* pred2  = (const float*)d_in[1];
  const float* tgt1   = (const float*)d_in[2];
  const float* tgt2   = (const float*)d_in[3];
  const float* coord1 = (const float*)d_in[4];
  const float* coord2 = (const float*)d_in[5];

  char* ws = (char*)d_ws;
  _Float16* q1p = (_Float16*)(ws + 0 * PACK_BYTES);
  _Float16* q2p = (_Float16*)(ws + 1 * PACK_BYTES);
  _Float16* k1p = (_Float16*)(ws + 2 * PACK_BYTES);
  _Float16* k2p = (_Float16*)(ws + 3 * PACK_BYTES);
  float* c1x = (float*)(ws + 4 * PACK_BYTES);
  float* c1y = c1x + CENT_ELEMS;
  float* c2x = c1y + CENT_ELEMS;
  float* c2y = c2x + CENT_ELEMS;
  float* r2  = c2y + CENT_ELEMS;
  float* acc = r2 + N_;

  dim3 nb(HW_ / 32, N_, 4);
  dim3 nt(32, 8);
  norm_pack_kernel<<<nb, nt, 0, stream>>>(pred1, pred2, tgt1, tgt2,
                                          q1p, q2p, k1p, k2p);

  setup_kernel<<<(N_ * HW_ + 255) / 256, 256, 0, stream>>>(
      coord1, coord2, c1x, c1y, c2x, c2y, r2, acc);

  dim3 gb((REGIONS_ + 7) / 8, N_, 2);   // 301 x 8 x 2 blocks, 8 waves each
  wmma_loss_kernel<<<gb, 256, 0, stream>>>(q1p, q2p, k1p, k2p,
                                           c1x, c1y, c2x, c2y, r2, acc);

  finalize_kernel<<<1, 32, 0, stream>>>(acc, (float*)d_out);
}
